// LCloss_17729624998436
// MI455X (gfx1250) — compile-verified, matrix operand order checked by reading, not run on hardware
//
#include <hip/hip_runtime.h>
#include <hip/hip_bf16.h>

// ---------------------------------------------------------------------------
// Problem constants (from reference): B=4, N=M=4096, D=128, temp=10
// ---------------------------------------------------------------------------
#define BATCH 4
#define NN    4096
#define DD    128
#define BN    128          // rows (n) per block
#define BM    64           // cols (m) per block iteration
#define LDA   136          // LDS row stride in bf16 elements (padded)
#define TEMP_INV 0.1f
#define CLIP_MAX 92.103403719761827f   // -log(1e-40)

typedef __attribute__((ext_vector_type(16))) __bf16 v16bf;
typedef __attribute__((ext_vector_type(8)))  float  v8f;
typedef __attribute__((ext_vector_type(4)))  unsigned v4u;
typedef __attribute__((ext_vector_type(4)))  int      v4i;
typedef __attribute__((ext_vector_type(8)))  int      v8i;

// Tensor Data Mover availability (arity differs across toolchains)
#if defined(__has_builtin)
#  if __has_builtin(__builtin_amdgcn_tensor_load_to_lds) && \
      __has_builtin(__builtin_amdgcn_s_wait_tensorcnt)
#    define HAVE_TDM 1
#  endif
#endif
#ifndef HAVE_TDM
#  define HAVE_TDM 0
#endif

// Map float -> monotonically ordered unsigned (for packed argmax via atomicMax)
__device__ __forceinline__ unsigned f2ord(float f) {
  unsigned u = __float_as_uint(f);
  return (u & 0x80000000u) ? ~u : (u | 0x80000000u);
}

// Load one 16x32 bf16 A/B fragment from LDS. `p` already points at
// (row-with-lane)*LDA + k_chunk_base. Lane-half g selects the K sub-group.
// ISA layout: VGPR v holds K-pair ( ((v&4)<<2) + g*8 + (v&3)*2 ).
__device__ __forceinline__ v16bf load_frag(const __bf16* p, int g) {
  v16bf f;
#pragma unroll
  for (int v = 0; v < 8; ++v) {
    int k = ((v & 4) << 2) + g * 8 + (v & 3) * 2;
    f[2 * v]     = p[k];
    f[2 * v + 1] = p[k + 1];
  }
  return f;
}

// ---------------------------------------------------------------------------
// Kernel 1: inv row norms of f_s + init packed argmax array
// ---------------------------------------------------------------------------
__global__ void __launch_bounds__(256)
norms_init_kernel(const float* __restrict__ f_s,
                  float* __restrict__ inv_ns,
                  unsigned long long* __restrict__ colmax) {
  int i = blockIdx.x * 256 + threadIdx.x;          // row id in [0, B*N)
  const float4* p = (const float4*)(f_s + (size_t)i * DD);
  float s = 0.0f;
#pragma unroll
  for (int q = 0; q < DD / 4; ++q) {
    float4 v = p[q];
    s += v.x * v.x + v.y * v.y + v.z * v.z + v.w * v.w;
  }
  inv_ns[i] = 1.0f / fmaxf(sqrtf(s), 1e-20f);
  colmax[i] = 0ull;                                // below any real packed value
}

// ---------------------------------------------------------------------------
// Kernel 2: fused batched GEMM (split-bf16 WMMA) + row sum-exp + col argmax
// grid = (NN/BN, BATCH), block = 256 threads (8 waves, 4x2 wave grid)
// B tiles are DMA'd into LDS by the Tensor Data Mover (double-buffered),
// overlapped with WMMA compute of the current tile.
// ---------------------------------------------------------------------------
__global__ void __launch_bounds__(256)
lc_gemm_kernel(const float* __restrict__ f_s,
               const float* __restrict__ f_t,
               const float* __restrict__ inv_ns,
               float* __restrict__ all_exp,
               unsigned long long* __restrict__ colmax) {
  const int b    = blockIdx.y;
  const int n0   = blockIdx.x * BN;
  const int tid  = threadIdx.x;
  const int lane = tid & 31;
  const int g    = lane >> 4;          // lane half
  const int ln   = lane & 15;
  const int w    = tid >> 5;
  const int wr   = w >> 1;             // wave row 0..3  (32 rows each)
  const int wc   = w & 1;              // wave col 0..1  (32 cols each)

  const float* fs_b = f_s + (size_t)b * NN * DD;
  const float* ft_b = f_t + (size_t)b * NN * DD;

  __shared__ __bf16 As_hi[BN * LDA], As_lo[BN * LDA];
  __shared__ __bf16 Bs_hi[BM * LDA], Bs_lo[BM * LDA];
  __shared__ float  rowsum[BN];
#if HAVE_TDM
  __shared__ float  stage[2][BM * DD];          // 2 x 32KB f32 DMA staging

  // Build a 1D D# (each B tile = contiguous BM*DD floats) and issue a TDM load.
  auto issue_tdm = [&](int m0s, unsigned lds_byte_off) {
    unsigned long long ga =
        (unsigned long long)(size_t)(ft_b + (size_t)m0s * DD);
    v4u g0;
    g0[0] = 1u;                                  // count=1, user mode
    g0[1] = lds_byte_off;                        // lds_addr
    g0[2] = (unsigned)(ga & 0xffffffffu);        // global_addr[31:0]
    g0[3] = (unsigned)((ga >> 32) & 0x01ffffffu) // global_addr[56:32]
            | (2u << 30);                        // type = 2 ("image")
    v8i g1;
    g1[0] = 0x00020000;                          // data_size = 4 bytes
    g1[1] = (int)(((BM * DD) & 0xffff) << 16);   // tensor_dim0[15:0]
    g1[2] = (int)(((BM * DD) >> 16) & 0xffff)    // tensor_dim0[31:16]
            | (1 << 16);                         // tensor_dim1 = 1
    g1[3] = (int)((unsigned)(BM * DD) << 16);    // tile_dim0 = BM*DD
    g1[4] = 1;                                   // tile_dim1 = 1, tile_dim2 = 0
    g1[5] = BM * DD;                             // tensor_dim0_stride[31:0]
    g1[6] = 0;
    g1[7] = 0;
    v4i gz = {0, 0, 0, 0};
#if __clang_major__ >= 23
    v8i gz8 = {0, 0, 0, 0, 0, 0, 0, 0};
    __builtin_amdgcn_tensor_load_to_lds(g0, g1, gz, gz, gz8, 0);
#else
    __builtin_amdgcn_tensor_load_to_lds(g0, g1, gz, gz, 0);
#endif
  };

  const unsigned lds_stage0 = (unsigned)(size_t)(void*)&stage[0][0];
  const unsigned lds_stage1 = (unsigned)(size_t)(void*)&stage[1][0];
  if (w == 0) issue_tdm(0, lds_stage0);          // prefetch tile 0
#endif

  // ---- stage A tile (128x128 f32 -> bf16 hi/lo) ----
#pragma unroll
  for (int q = 0; q < 16; ++q) {
    int f4 = tid + 256 * q;            // coalesced float4 index
    int r  = f4 >> 5;                  // 32 float4 per row
    int c4 = f4 & 31;
    float4 v = ((const float4*)(fs_b + (size_t)(n0 + r) * DD))[c4];
    float vv[4] = {v.x, v.y, v.z, v.w};
#pragma unroll
    for (int e = 0; e < 4; ++e) {
      __bf16 h = (__bf16)vv[e];
      As_hi[r * LDA + c4 * 4 + e] = h;
      As_lo[r * LDA + c4 * 4 + e] = (__bf16)(vv[e] - (float)h);
    }
  }
  if (tid < BN) rowsum[tid] = 0.0f;

  // per-lane resident row data: rows  n0 + wr*32 + sr*16 + j + 8*g
  float invn[2][8], rexp[2][8];
#pragma unroll
  for (int sr = 0; sr < 2; ++sr)
#pragma unroll
    for (int j = 0; j < 8; ++j) {
      invn[sr][j] = inv_ns[(size_t)b * NN + n0 + wr * 32 + sr * 16 + j + 8 * g];
      rexp[sr][j] = 0.0f;
    }

  for (int it = 0; it < NN / BM; ++it) {
    const int m0 = it * BM;
#if HAVE_TDM
    if (w == 0) __builtin_amdgcn_s_wait_tensorcnt(0);   // tile `it` arrived
    __syncthreads();   // publish DMA data; close previous tile's Bs readers
    // ---- convert staged f32 tile -> bf16 hi/lo (LDS -> LDS) ----
    {
      const float4* sb = (const float4*)&stage[it & 1][0];
#pragma unroll
      for (int q = 0; q < 8; ++q) {
        int f4 = tid + 256 * q;
        int r  = f4 >> 5;
        int c4 = f4 & 31;
        float4 v = sb[f4];
        float vv[4] = {v.x, v.y, v.z, v.w};
#pragma unroll
        for (int e = 0; e < 4; ++e) {
          __bf16 h = (__bf16)vv[e];
          Bs_hi[r * LDA + c4 * 4 + e] = h;
          Bs_lo[r * LDA + c4 * 4 + e] = (__bf16)(vv[e] - (float)h);
        }
      }
    }
    // overlap: DMA next tile into the other staging buffer during compute
    if (w == 0 && it + 1 < NN / BM)
      issue_tdm(m0 + BM, (it & 1) ? lds_stage0 : lds_stage1);
    __syncthreads();
#else
    __syncthreads();
    // ---- stage B tile (64x128 f32 -> bf16 hi/lo) from global ----
#pragma unroll
    for (int q = 0; q < 8; ++q) {
      int f4 = tid + 256 * q;
      int r  = f4 >> 5;
      int c4 = f4 & 31;
      float4 v = ((const float4*)(ft_b + (size_t)(m0 + r) * DD))[c4];
      float vv[4] = {v.x, v.y, v.z, v.w};
#pragma unroll
      for (int e = 0; e < 4; ++e) {
        __bf16 h = (__bf16)vv[e];
        Bs_hi[r * LDA + c4 * 4 + e] = h;
        Bs_lo[r * LDA + c4 * 4 + e] = (__bf16)(vv[e] - (float)h);
      }
    }
    if (m0 + BM < NN)   // pull next B tile toward L2/L0 (global_prefetch_b8)
      __builtin_prefetch(ft_b + (size_t)(m0 + BM + (tid & 63)) * DD, 0, 0);
    __syncthreads();
#endif

    // ---- WMMA: 2x2 subtiles, K=128 in 4 chunks, 3 products (hi/lo split) ----
    const v8f vzero = {0, 0, 0, 0, 0, 0, 0, 0};
    v8f acc[2][2];
    acc[0][0] = vzero; acc[0][1] = vzero; acc[1][0] = vzero; acc[1][1] = vzero;
#pragma unroll
    for (int kc = 0; kc < 4; ++kc) {
      v16bf ah[2], al[2], bh[2], bl[2];
#pragma unroll
      for (int sr = 0; sr < 2; ++sr) {
        const int ro = (wr * 32 + sr * 16 + ln) * LDA + kc * 32;
        ah[sr] = load_frag(&As_hi[ro], g);
        al[sr] = load_frag(&As_lo[ro], g);
      }
#pragma unroll
      for (int sc = 0; sc < 2; ++sc) {
        const int co = (wc * 32 + sc * 16 + ln) * LDA + kc * 32;
        bh[sc] = load_frag(&Bs_hi[co], g);
        bl[sc] = load_frag(&Bs_lo[co], g);
      }
#pragma unroll
      for (int sr = 0; sr < 2; ++sr)
#pragma unroll
        for (int sc = 0; sc < 2; ++sc) {
          acc[sr][sc] = __builtin_amdgcn_wmma_f32_16x16x32_bf16(
              false, ah[sr], false, bh[sc], (short)0, acc[sr][sc], false, false);
          acc[sr][sc] = __builtin_amdgcn_wmma_f32_16x16x32_bf16(
              false, ah[sr], false, bl[sc], (short)0, acc[sr][sc], false, false);
          acc[sr][sc] = __builtin_amdgcn_wmma_f32_16x16x32_bf16(
              false, al[sr], false, bh[sc], (short)0, acc[sr][sc], false, false);
        }
    }

    // ---- epilogue A: accumulate sum_m exp(dot/temp) per row (in registers) ----
#pragma unroll
    for (int sr = 0; sr < 2; ++sr)
#pragma unroll
      for (int sc = 0; sc < 2; ++sc)
#pragma unroll
        for (int j = 0; j < 8; ++j)
          rexp[sr][j] += __expf(acc[sr][sc][j] * TEMP_INV);

    // ---- epilogue B: column argmax of dot/||f_s[n]|| (packed value|index) ----
#pragma unroll
    for (int sc = 0; sc < 2; ++sc) {
      unsigned long long best = 0ull;
#pragma unroll
      for (int sr = 0; sr < 2; ++sr)
#pragma unroll
        for (int j = 0; j < 8; ++j) {
          float c = acc[sr][sc][j] * invn[sr][j];
          unsigned long long pk =
              ((unsigned long long)f2ord(c) << 32) |
              (unsigned)(n0 + wr * 32 + sr * 16 + j + 8 * g);
          best = best > pk ? best : pk;
        }
      unsigned long long o = __shfl_xor(best, 16);  // merge lane halves (same cols)
      best = best > o ? best : o;
      if (g == 0)
        atomicMax(&colmax[(size_t)b * NN + m0 + wc * 32 + sc * 16 + ln], best);
    }
  }

  // ---- finish row sums: reduce 16 lanes, combine 2 col-waves via LDS atomic ----
#pragma unroll
  for (int sr = 0; sr < 2; ++sr)
#pragma unroll
    for (int j = 0; j < 8; ++j) {
      float s = rexp[sr][j];
      s += __shfl_xor(s, 1); s += __shfl_xor(s, 2);
      s += __shfl_xor(s, 4); s += __shfl_xor(s, 8);
      if (ln == 0) atomicAdd(&rowsum[wr * 32 + sr * 16 + j + 8 * g], s);
    }
  __syncthreads();
  if (tid < BN) all_exp[(size_t)b * NN + n0 + tid] = rowsum[tid];
}

// ---------------------------------------------------------------------------
// Kernel 3: per-row term = min( log(all_exp) - <f_s[n], f_trans[idx]>/temp, 92.1 )
// one wave per row; per-block partial sums (deterministic)
// ---------------------------------------------------------------------------
__global__ void __launch_bounds__(256)
finalize_kernel(const float* __restrict__ f_s,
                const float* __restrict__ f_t,
                const float* __restrict__ all_exp,
                const unsigned long long* __restrict__ colmax,
                float* __restrict__ partials) {
  const int wv   = threadIdx.x >> 5;
  const int lane = threadIdx.x & 31;
  const int row  = blockIdx.x * 8 + wv;            // global (b,n) flat
  const int b    = row >> 12;                      // N = 4096
  const unsigned idx = (unsigned)(colmax[row] & 0xffffffffu);

  float4 a = ((const float4*)(f_s + (size_t)row * DD))[lane];
  float4 t = ((const float4*)(f_t + ((size_t)b * NN + idx) * DD))[lane];
  float d = a.x * t.x + a.y * t.y + a.z * t.z + a.w * t.w;
  d += __shfl_xor(d, 1);  d += __shfl_xor(d, 2);
  d += __shfl_xor(d, 4);  d += __shfl_xor(d, 8);  d += __shfl_xor(d, 16);

  float term = logf(all_exp[row]) - d * TEMP_INV;
  term = fminf(term, CLIP_MAX);

  __shared__ float sm[8];
  if (lane == 0) sm[wv] = term;
  __syncthreads();
  if (threadIdx.x == 0) {
    float s = 0.0f;
#pragma unroll
    for (int k = 0; k < 8; ++k) s += sm[k];
    partials[blockIdx.x] = s;
  }
}

__global__ void __launch_bounds__(256)
reduce_kernel(const float* __restrict__ partials, float* __restrict__ out) {
  float s = 0.0f;
  for (int i = threadIdx.x; i < 2048; i += 256) s += partials[i];
  __shared__ float sm[256];
  sm[threadIdx.x] = s;
  __syncthreads();
  for (int off = 128; off > 0; off >>= 1) {
    if (threadIdx.x < off) sm[threadIdx.x] += sm[threadIdx.x + off];
    __syncthreads();
  }
  if (threadIdx.x == 0) out[0] = sm[0] * (1.0f / (float)(BATCH * NN));
}

// ---------------------------------------------------------------------------
extern "C" void kernel_launch(void* const* d_in, const int* in_sizes, int n_in,
                              void* d_out, int out_size, void* d_ws, size_t ws_size,
                              hipStream_t stream) {
  const float* f_s = (const float*)d_in[0];
  const float* f_t = (const float*)d_in[1];

  float* ws = (float*)d_ws;
  float* inv_ns  = ws;                                    //  64 KB
  float* all_exp = ws + BATCH * NN;                       //  64 KB
  unsigned long long* colmax =
      (unsigned long long*)(ws + 2 * BATCH * NN);         // 128 KB (8B aligned)
  float* partials = (float*)(colmax + BATCH * NN);        //   8 KB

  norms_init_kernel<<<(BATCH * NN) / 256, 256, 0, stream>>>(f_s, inv_ns, colmax);

  dim3 grid(NN / BN, BATCH);
  lc_gemm_kernel<<<grid, 256, 0, stream>>>(f_s, f_t, inv_ns, all_exp, colmax);

  finalize_kernel<<<(BATCH * NN) / 8, 256, 0, stream>>>(f_s, f_t, all_exp,
                                                        colmax, partials);
  reduce_kernel<<<1, 256, 0, stream>>>(partials, (float*)d_out);
}
